// PoHBlock_24275155157517
// MI455X (gfx1250) — matrix operand order repaired and checked by reading, not compiled, not run on hardware
//
#include <hip/hip_runtime.h>
#include <hip/hip_bf16.h>

#define DEV __device__ __forceinline__

typedef __bf16 bf16_t;
typedef __attribute__((ext_vector_type(16))) __bf16 v16bf;
typedef __attribute__((ext_vector_type(8)))  __bf16 v8bf;
typedef __attribute__((ext_vector_type(8)))  float  v8f;

static constexpr int Bb   = 2;
static constexpr int Ss   = 2048;
static constexpr int Dm   = 512;
static constexpr int Hh   = 8;
static constexpr int Dh   = 64;
static constexpr int Dff  = 2048;
static constexpr int Mrows = Bb * Ss;   // 4096

// ---------- scalar conversion helpers ----------
DEV bf16_t f2bf(float f) {
  unsigned u; __builtin_memcpy(&u, &f, 4);
  unsigned r = u + 0x7FFFu + ((u >> 16) & 1u);   // round-to-nearest-even
  unsigned short h = (unsigned short)(r >> 16);
  bf16_t o; __builtin_memcpy(&o, &h, 2);
  return o;
}

// flat pointer -> 32-bit LDS byte offset (aperture low bits are the wave-relative
// LDS address per the CDNA5 aperture mapping)
DEV unsigned lds_u32(const void* p) { return (unsigned)(unsigned long long)p; }

// async-copy 32 bytes global -> LDS (two b128 transfers, tracked by ASYNCcnt)
DEV void async_copy32(unsigned lds_dst, unsigned long long gsrc) {
  asm volatile(
      "global_load_async_to_lds_b128 %0, %1, off\n\t"
      "global_load_async_to_lds_b128 %0, %1, off offset:16"
      :: "v"(lds_dst), "v"(gsrc) : "memory");
}
DEV void async_wait0() {
  asm volatile("s_wait_asynccnt 0" ::: "memory");
}

// ---------- WMMA fragment helpers (16x16x32 bf16, contiguous k order used for both A and B) ----------
// A (16x32, MxK): lane L -> M=L&15, half h=L>>4; elems 0..15 = K(16h .. 16h+15).
//                 Source row-major, K contiguous.
DEV v16bf load_a_frag(const bf16_t* __restrict__ base, int ldk, int lane) {
  int m = lane & 15, h = lane >> 4;
  const bf16_t* p = base + (long)m * ldk + 16 * h;
  v8bf lo = *(const v8bf*)p;
  v8bf hi = *(const v8bf*)(p + 8);
  return __builtin_shufflevector(lo, hi, 0,1,2,3,4,5,6,7,8,9,10,11,12,13,14,15);
}

// B (32x16, KxN): lane L -> N=L&15, half h=L>>4; elems 0..15 = K(16h .. 16h+15).
// Source stored [N][K] (K contiguous, stride ldk).
DEV v16bf load_b_frag(const bf16_t* __restrict__ baseT, int ldk, int lane) {
  int n = lane & 15, h = lane >> 4;
  const bf16_t* p = baseT + (long)n * ldk + 16 * h;
  v8bf lo = *(const v8bf*)p;
  v8bf hi = *(const v8bf*)(p + 8);
  return __builtin_shufflevector(lo, hi, 0,1,2,3,4,5,6,7,8,9,10,11,12,13,14,15);
}

DEV v8f wmma_bf16(v16bf a, v16bf b, v8f c) {
  return __builtin_amdgcn_wmma_f32_16x16x32_bf16(false, a, false, b, (short)0, c, false, false);
}

// ---------- weight / activation prep ----------
__global__ void prep_qkv_w(const float* __restrict__ W, bf16_t* __restrict__ out) {
  int idx = blockIdx.x * blockDim.x + threadIdx.x;     // H*512*64 = 262144
  int e = idx & 63, d = (idx >> 6) & 511, h = idx >> 15;
  out[(h * 64 + e) * Dm + d] = f2bf(W[idx]);           // -> [N=h*64+e][K=d]
}

__global__ void prep_mat_w(const float* __restrict__ W, bf16_t* __restrict__ out,
                           int K, int N) {             // W row-major [K][N] -> [N][K]
  int idx = blockIdx.x * blockDim.x + threadIdx.x;
  if (idx >= K * N) return;
  int n = idx % N, kk = idx / N;
  out[(long)n * K + kk] = f2bf(W[idx]);
}

__global__ void cast_z(const float* __restrict__ z, float* __restrict__ zc,
                       bf16_t* __restrict__ zb, int n) {
  int idx = blockIdx.x * blockDim.x + threadIdx.x;
  if (idx < n) { float v = z[idx]; zc[idx] = v; zb[idx] = f2bf(v); }
}

// ---------- QKV projection: [4096x512] @ [512x512] (+bias), custom output layout ----------
// vmode 0: out[(b*H+h)][s][e]   (q and k)
// vmode 1: out[(b*H+h)][e][s]   (v transposed, so PV B-frags are contiguous)
__global__ void proj_kernel(const bf16_t* __restrict__ A, const bf16_t* __restrict__ Wt,
                            const float* __restrict__ bias, bf16_t* __restrict__ out,
                            int vmode) {
  int lane = threadIdx.x & 31, wid = threadIdx.x >> 5;
  int row0 = (blockIdx.y * 4 + wid) * 16;
  int n0 = blockIdx.x * 64;
  v8f c[4] = { {}, {}, {}, {} };
  for (int k = 0; k < Dm; k += 32) {
    v16bf a = load_a_frag(A + (long)row0 * Dm + k, Dm, lane);
#pragma unroll
    for (int f = 0; f < 4; f++) {
      v16bf b = load_b_frag(Wt + (long)(n0 + f * 16) * Dm + k, Dm, lane);
      c[f] = wmma_bf16(a, b, c[f]);
    }
  }
  int nl = lane & 15, hl = lane >> 4;
#pragma unroll
  for (int f = 0; f < 4; f++) {
    int n = n0 + f * 16 + nl;
    float bv = bias[n];
    int hh = n >> 6, e = n & 63;
#pragma unroll
    for (int r = 0; r < 8; r++) {
      int m = row0 + r + 8 * hl;
      int b_ = m >> 11, s = m & 2047;
      float val = c[f][r] + bv;
      long idx = (vmode == 0)
                   ? ((long)((b_ * Hh + hh) * Ss + s) * Dh + e)
                   : ((long)((b_ * Hh + hh) * Dh + e) * Ss + s);
      out[idx] = f2bf(val);
    }
  }
}

// ---------- flash attention ----------
// Block = 4 waves, each wave owns a 16-query tile; K/V chunk (32 kv positions) is
// async-staged into LDS once per block and shared by all 4 waves.
__global__ void flash_kernel(const bf16_t* __restrict__ q, const bf16_t* __restrict__ k,
                             const bf16_t* __restrict__ vT, bf16_t* __restrict__ attnout) {
  __shared__ alignas(32) bf16_t kt[32 * 64];        // [kv_local][e]   4KB
  __shared__ alignas(32) bf16_t vt[64 * 32];        // [e][kv_local]   4KB
  __shared__ alignas(32) bf16_t lds_p[4][16 * 32];  // per-wave P transpose buffer
  int t = threadIdx.x;
  int lane = t & 31, wid = t >> 5;
  int bh = blockIdx.y;                        // 0..15  (b*8+h)
  int qr = (blockIdx.x * 4 + wid) * 16;       // query row base
  const bf16_t* qb = q + ((long)bh * Ss + qr) * Dh;
  const bf16_t* kb = k + (long)bh * Ss * Dh;
  const bf16_t* vb = vT + (long)bh * Dh * Ss;

  v16bf aq0 = load_a_frag(qb, Dh, lane);       // K = 0..31 of d_head
  v16bf aq1 = load_a_frag(qb + 32, Dh, lane);  // K = 32..63
  v8f acc[4] = { {}, {}, {}, {} };
  float mrun[8], lsum[8];
#pragma unroll
  for (int r = 0; r < 8; r++) { mrun[r] = -3.0e38f; lsum[r] = 0.f; }

  bf16_t* lp = lds_p[wid];
  const float scale = 0.125f;                  // 1/sqrt(64)
  int nl = lane & 15, hl = lane >> 4;

  // per-thread staging addresses (fixed across iterations)
  unsigned k_dst = lds_u32(kt) + (unsigned)(t * 32);                 // contiguous 4KB tile
  int vrow = t >> 1, vpart = t & 1;                                  // 64 rows x two 32B halves
  unsigned v_dst = lds_u32(vt) + (unsigned)((vrow * 32 + vpart * 16) * 2);

  for (int kv = 0; kv < Ss; kv += 32) {
    // ---- async stage K/V chunk into LDS ----
    unsigned long long k_src = (unsigned long long)(const void*)(kb + (long)kv * Dh)
                               + (unsigned)(t * 32);
    unsigned long long v_src = (unsigned long long)(const void*)
                               (vb + (long)vrow * Ss + kv + vpart * 16);
    async_copy32(k_dst, k_src);
    async_copy32(v_dst, v_src);
    async_wait0();
    __syncthreads();

    // ---- scores = q @ k^T ----
    v8f sc[2];
#pragma unroll
    for (int nh = 0; nh < 2; nh++) {
      const bf16_t* kcol = kt + (nh * 16) * Dh;    // LDS: [kv_local][e]
      v16bf b0 = load_b_frag(kcol, Dh, lane);
      v16bf b1 = load_b_frag(kcol + 32, Dh, lane);
      v8f s = {};
      s = wmma_bf16(aq0, b0, s);
      s = wmma_bf16(aq1, b1, s);
      sc[nh] = s;
    }
    float cm[8];
#pragma unroll
    for (int r = 0; r < 8; r++) {
      sc[0][r] *= scale; sc[1][r] *= scale;
      cm[r] = fmaxf(sc[0][r], sc[1][r]);
    }
#pragma unroll
    for (int mask = 1; mask <= 8; mask <<= 1) {
#pragma unroll
      for (int r = 0; r < 8; r++)
        cm[r] = fmaxf(cm[r], __shfl_xor(cm[r], mask, 32));
    }
    float sf[8], rs[8];
    v8f p0, p1;
#pragma unroll
    for (int r = 0; r < 8; r++) {
      float mn = fmaxf(mrun[r], cm[r]);
      sf[r] = __expf(mrun[r] - mn);
      mrun[r] = mn;
      p0[r] = __expf(sc[0][r] - mn);
      p1[r] = __expf(sc[1][r] - mn);
      rs[r] = p0[r] + p1[r];
    }
#pragma unroll
    for (int mask = 1; mask <= 8; mask <<= 1) {
#pragma unroll
      for (int r = 0; r < 8; r++)
        rs[r] += __shfl_xor(rs[r], mask, 32);
    }
#pragma unroll
    for (int r = 0; r < 8; r++) lsum[r] = lsum[r] * sf[r] + rs[r];
#pragma unroll
    for (int f = 0; f < 4; f++) {
#pragma unroll
      for (int r = 0; r < 8; r++) acc[f][r] *= sf[r];
    }
    // C-layout (lane=N, VGPR=row) -> LDS row-major 16x32, reload as A-frag
#pragma unroll
    for (int r = 0; r < 8; r++) {
      int row = r + 8 * hl;
      lp[row * 32 + nl]      = f2bf(p0[r]);
      lp[row * 32 + 16 + nl] = f2bf(p1[r]);
    }
    __syncthreads();
    v16bf pa = load_a_frag(lp, 32, lane);
#pragma unroll
    for (int f = 0; f < 4; f++) {
      v16bf bvf = load_b_frag(vt + (f * 16) * 32, 32, lane);  // LDS: [e][kv_local]
      acc[f] = wmma_bf16(pa, bvf, acc[f]);
    }
    __syncthreads();   // all reads of kt/vt done before next chunk's staging
  }
  int b_ = bh >> 3, hcol = (bh & 7) * Dh;
#pragma unroll
  for (int r = 0; r < 8; r++) {
    float inv = 1.f / lsum[r];
    int s = qr + r + 8 * hl;
#pragma unroll
    for (int f = 0; f < 4; f++)
      attnout[(long)(b_ * Ss + s) * Dm + hcol + f * 16 + nl] = f2bf(acc[f][r] * inv);
  }
}

// ---------- generic GEMM: C[MxN] = A_bf[MxK] @ Wt[N][K] + bias (opt. ReLU) ----------
// Weight tile (64 cols x 32 k, shared by all 4 waves of the block) is async-staged
// into LDS with double buffering so the DMA overlaps the WMMAs.
__global__ void gemm_kernel(const bf16_t* __restrict__ A, const bf16_t* __restrict__ Wt,
                            const float* __restrict__ bias, float* __restrict__ outf,
                            bf16_t* __restrict__ outbf, int N, int K, int relu) {
  __shared__ alignas(32) bf16_t wtile[2][64 * 32];   // [n_local][k_local], 4KB per buffer
  int t = threadIdx.x;
  int lane = t & 31, wid = t >> 5;
  int row0 = blockIdx.y * 64 + wid * 16;
  int n0 = blockIdx.x * 64;

  int nrow = t >> 1, npart = t & 1;                  // 64 rows x two 32B halves
  unsigned dstoff = (unsigned)((nrow * 32 + npart * 16) * 2);
  unsigned dst[2] = { lds_u32(&wtile[0][0]) + dstoff, lds_u32(&wtile[1][0]) + dstoff };
  const bf16_t* wsrc = Wt + (long)(n0 + nrow) * K + npart * 16;

  // prologue: stage k=0 tile
  async_copy32(dst[0], (unsigned long long)(const void*)wsrc);
  async_wait0();
  __syncthreads();

  v8f c[4] = { {}, {}, {}, {} };
  int buf = 0;
  for (int k = 0; k < K; k += 32, buf ^= 1) {
    if (k + 32 < K)   // stage next tile while computing on current one
      async_copy32(dst[buf ^ 1], (unsigned long long)(const void*)(wsrc + k + 32));
    __builtin_prefetch(A + (long)row0 * K + k + 2048, 0, 0);
    v16bf a = load_a_frag(A + (long)row0 * K + k, K, lane);
#pragma unroll
    for (int f = 0; f < 4; f++) {
      v16bf b = load_b_frag(&wtile[buf][(f * 16) * 32], 32, lane);
      c[f] = wmma_bf16(a, b, c[f]);
    }
    async_wait0();       // next buffer landed (overlapped with the WMMAs above)
    __syncthreads();     // and everyone is done reading the current buffer
  }
  int nl = lane & 15, hl = lane >> 4;
#pragma unroll
  for (int f = 0; f < 4; f++) {
    int n = n0 + f * 16 + nl;
    float bv = bias[n];
#pragma unroll
    for (int r = 0; r < 8; r++) {
      int m = row0 + r + 8 * hl;
      float val = c[f][r] + bv;
      if (relu) val = fmaxf(val, 0.f);
      long idx = (long)m * N + n;
      if (outf)  outf[idx]  = val;
      if (outbf) outbf[idx] = f2bf(val);
    }
  }
}

// ---------- residual add + LayerNorm over D=512 (one block per row) ----------
__global__ void resid_ln_kernel(const float* __restrict__ resid, const float* __restrict__ x,
                                const float* __restrict__ g, const float* __restrict__ be,
                                float* __restrict__ outf, bf16_t* __restrict__ outbf) {
  int row = blockIdx.x;
  int t = threadIdx.x;                          // 128 threads
  const float* rp = resid + (long)row * Dm;
  const float* xp = x + (long)row * Dm;
  float v[4]; float s = 0.f, s2 = 0.f;
#pragma unroll
  for (int i = 0; i < 4; i++) {
    int c = t + i * 128;
    float val = rp[c] + xp[c];
    v[i] = val; s += val; s2 += val * val;
  }
#pragma unroll
  for (int mask = 1; mask < 32; mask <<= 1) {
    s  += __shfl_xor(s,  mask, 32);
    s2 += __shfl_xor(s2, mask, 32);
  }
  __shared__ float red[8];
  int wid = t >> 5, lane = t & 31;
  if (lane == 0) { red[wid] = s; red[4 + wid] = s2; }
  __syncthreads();
  float ts  = red[0] + red[1] + red[2] + red[3];
  float ts2 = red[4] + red[5] + red[6] + red[7];
  float mu  = ts * (1.f / Dm);
  float var = ts2 * (1.f / Dm) - mu * mu;
  float rstd = rsqrtf(var + 1e-5f);
#pragma unroll
  for (int i = 0; i < 4; i++) {
    int c = t + i * 128;
    float val = (v[i] - mu) * rstd * g[c] + be[c];
    outf[(long)row * Dm + c]  = val;
    outbf[(long)row * Dm + c] = f2bf(val);
  }
}

// ---------- host launcher ----------
extern "C" void kernel_launch(void* const* d_in, const int* in_sizes, int n_in,
                              void* d_out, int out_size, void* d_ws, size_t ws_size,
                              hipStream_t stream) {
  (void)in_sizes; (void)n_in; (void)out_size; (void)ws_size;
  const float* z   = (const float*)d_in[0];
  const float* Wq  = (const float*)d_in[1];
  const float* bq  = (const float*)d_in[2];
  const float* Wk  = (const float*)d_in[3];
  const float* bk  = (const float*)d_in[4];
  const float* Wv  = (const float*)d_in[5];
  const float* bv  = (const float*)d_in[6];
  const float* Wo  = (const float*)d_in[7];
  const float* bo  = (const float*)d_in[8];
  const float* W1  = (const float*)d_in[9];
  const float* b1  = (const float*)d_in[10];
  const float* W2  = (const float*)d_in[11];
  const float* b2  = (const float*)d_in[12];
  const float* g1  = (const float*)d_in[13];
  const float* be1 = (const float*)d_in[14];
  const float* g2  = (const float*)d_in[15];
  const float* be2 = (const float*)d_in[16];
  float* outp = (float*)d_out;

  char* w = (char*)d_ws;
  size_t off = 0;
  auto alloc = [&](size_t bytes) -> void* {
    void* p = w + off;
    off += (bytes + 255) & ~(size_t)255;
    return p;
  };
  bf16_t* Wq_t = (bf16_t*)alloc((size_t)Dm * Dm * 2);
  bf16_t* Wk_t = (bf16_t*)alloc((size_t)Dm * Dm * 2);
  bf16_t* Wv_t = (bf16_t*)alloc((size_t)Dm * Dm * 2);
  bf16_t* Wo_t = (bf16_t*)alloc((size_t)Dm * Dm * 2);
  bf16_t* W1_t = (bf16_t*)alloc((size_t)Dm * Dff * 2);
  bf16_t* W2_t = (bf16_t*)alloc((size_t)Dff * Dm * 2);
  bf16_t* z_bf = (bf16_t*)alloc((size_t)Mrows * Dm * 2);
  float*  zcur = (float*)alloc((size_t)Mrows * Dm * 4);
  bf16_t* q_bf = (bf16_t*)alloc((size_t)Bb * Hh * Ss * Dh * 2);
  bf16_t* k_bf = (bf16_t*)alloc((size_t)Bb * Hh * Ss * Dh * 2);
  bf16_t* vT_bf = (bf16_t*)alloc((size_t)Bb * Hh * Dh * Ss * 2);
  bf16_t* attn_bf = (bf16_t*)alloc((size_t)Mrows * Dm * 2);
  float*  gemmout = (float*)alloc((size_t)Mrows * Dm * 4);
  float*  zln1    = (float*)alloc((size_t)Mrows * Dm * 4);
  bf16_t* zln1_bf = (bf16_t*)alloc((size_t)Mrows * Dm * 2);
  bf16_t* h1_bf   = (bf16_t*)alloc((size_t)Mrows * Dff * 2);

  // weight prep (bf16, [N][K])
  prep_qkv_w<<<1024, 256, 0, stream>>>(Wq, Wq_t);
  prep_qkv_w<<<1024, 256, 0, stream>>>(Wk, Wk_t);
  prep_qkv_w<<<1024, 256, 0, stream>>>(Wv, Wv_t);
  prep_mat_w<<<1024, 256, 0, stream>>>(Wo, Wo_t, Dm, Dm);
  prep_mat_w<<<4096, 256, 0, stream>>>(W1, W1_t, Dm, Dff);
  prep_mat_w<<<4096, 256, 0, stream>>>(W2, W2_t, Dff, Dm);
  cast_z<<<8192, 256, 0, stream>>>(z, zcur, z_bf, Mrows * Dm);

  for (int it = 0; it < 4; it++) {
    proj_kernel<<<dim3(8, 64), 128, 0, stream>>>(z_bf, Wq_t, bq, q_bf, 0);
    proj_kernel<<<dim3(8, 64), 128, 0, stream>>>(z_bf, Wk_t, bk, k_bf, 0);
    proj_kernel<<<dim3(8, 64), 128, 0, stream>>>(z_bf, Wv_t, bv, vT_bf, 1);
    flash_kernel<<<dim3(32, 16), 128, 0, stream>>>(q_bf, k_bf, vT_bf, attn_bf);
    gemm_kernel<<<dim3(8, 64), 128, 0, stream>>>(attn_bf, Wo_t, bo,
                                                 gemmout, (bf16_t*)nullptr, Dm, Dm, 0);
    resid_ln_kernel<<<Mrows, 128, 0, stream>>>(zcur, gemmout, g1, be1,
                                               (it == 3) ? outp : zln1, zln1_bf);
    if (it < 3) {
      gemm_kernel<<<dim3(32, 64), 128, 0, stream>>>(zln1_bf, W1_t, b1,
                                                    (float*)nullptr, h1_bf, Dff, Dm, 1);
      gemm_kernel<<<dim3(8, 64), 128, 0, stream>>>(h1_bf, W2_t, b2,
                                                   gemmout, (bf16_t*)nullptr, Dm, Dff, 0);
      resid_ln_kernel<<<Mrows, 128, 0, stream>>>(zln1, gemmout, g2, be2, zcur, z_bf);
    }
  }
}
